// CLAHEModule_10290741641714
// MI455X (gfx1250) — compile-verified, matrix operand order checked by reading, not run on hardware
//
#include <hip/hip_runtime.h>
#include <hip/hip_bf16.h>

typedef __attribute__((ext_vector_type(16))) _Float16 v16h;
typedef __attribute__((ext_vector_type(8)))  float    v8f;

#define TILE_G   8
#define BINS     256
#define TDIM     64            // tile is 64x64
#define TAREA    4096
#define CLIPV    32            // int(2.0*4096/256)
#define IMG_W    512
#define IMG_HW   (512*512)

// ---------------------------------------------------------------------------
// Kernel 1: one block per tile. Histogram (LDS atomics) -> clip/redistribute
// -> 256-bin prefix sum via two v_wmma_f32_16x16x32_f16 -> uint8 LUT.
// ---------------------------------------------------------------------------
__global__ void __launch_bounds__(256)
clahe_lut_kernel(const float* __restrict__ x, unsigned char* __restrict__ lut)
{
    __shared__ int   s_hist[BINS];
    __shared__ float s_histF[BINS];
    __shared__ float s_T[16];
    __shared__ int   s_excess;

    const int tid = threadIdx.x;
    const int t   = blockIdx.x;          // tile id, 0..6143
    const int bc  = t >> 6;              // image (b*3+c), 0..95
    const int tt  = t & 63;
    const int gy  = tt >> 3;
    const int gx  = tt & 7;

    s_hist[tid] = 0;
    if (tid == 0) s_excess = 0;
    __syncthreads();

    // --- histogram: 16 pixels per thread, coalesced rows of 64 ---
    const int base = bc * IMG_HW + (gy * TDIM) * IMG_W + gx * TDIM;
    #pragma unroll
    for (int i = 0; i < 16; ++i) {
        int p   = i * 256 + tid;         // 0..4095 within tile
        int row = p >> 6;
        int col = p & 63;
        float xv = x[base + row * IMG_W + col];
        int v = (int)rintf(fminf(fmaxf(xv, 0.f), 1.f) * 255.f);
        atomicAdd(&s_hist[v], 1);
    }
    __syncthreads();

    // --- clip + accumulate excess ---
    int h = s_hist[tid];
    int c = (h < CLIPV) ? h : CLIPV;
    atomicAdd(&s_excess, h - c);
    __syncthreads();

    // --- redistribute excess (OpenCV semantics) ---
    {
        int ex        = s_excess;
        int batch_add = ex >> 8;
        int residual  = ex & 255;
        int rmax      = (residual > 1) ? residual : 1;
        int step      = 256 / rmax; if (step < 1) step = 1;
        int extra     = ((tid % step) == 0 && (tid / step) < residual) ? 1 : 0;
        s_histF[tid]  = (float)(c + batch_add + extra);
    }
    __syncthreads();

    // --- row totals T[r] = sum of 16 bins of row r (needed for offsets) ---
    if (tid < 16) {
        float s = 0.f;
        #pragma unroll
        for (int k = 0; k < 16; ++k) s += s_histF[tid * 16 + k];
        s_T[tid] = s;
    }
    __syncthreads();

    // --- wave 0: 256-wide inclusive scan via WMMA ---
    // cdf(16x16) = Lstrict x Tbcast + (H x Uincl)
    if (tid < 32) {
        const int lane = tid;
        const int hi   = lane >> 4;      // 0 or 1
        const int mn   = lane & 15;      // M for A, N for B, N for C/D

        v16h a1, b1, a2, b2;
        #pragma unroll
        for (int e = 0; e < 16; ++e) {
            int k = ((e < 8) ? e : e + 8) + 8 * hi;   // ISA 16-bit A/B layout
            bool kv = (k < 16);
            a1[e] = kv ? (_Float16)s_histF[mn * 16 + k] : (_Float16)0.f; // H[m][k]
            b1[e] = (kv && k <= mn) ? (_Float16)1.f : (_Float16)0.f;     // U[k][n]
            a2[e] = (kv && k <  mn) ? (_Float16)1.f : (_Float16)0.f;     // Ls[m][k]
            b2[e] = kv ? (_Float16)s_T[k] : (_Float16)0.f;               // Tb[k][n]
        }

        v8f acc = {};
        // P = H x U
        acc = __builtin_amdgcn_wmma_f32_16x16x32_f16(
                  false, a1, false, b1, (short)0, acc, false, false);
        // cdf = Ls x Tb + P
        acc = __builtin_amdgcn_wmma_f32_16x16x32_f16(
                  false, a2, false, b2, (short)0, acc, false, false);

        const float scale = 255.f / 4096.f;   // (BINS-1)/tile_area, exact fp32
        unsigned char* lt = lut + t * BINS;
        #pragma unroll
        for (int v = 0; v < 8; ++v) {
            int m = v + 8 * hi;
            float l = rintf(acc[v] * scale);
            l = fminf(fmaxf(l, 0.f), 255.f);
            lt[m * 16 + mn] = (unsigned char)l;
        }
    }
}

// ---------------------------------------------------------------------------
// Kernel 2: bilinear LUT apply, float4-vectorized (4 pixels along W / thread)
// ---------------------------------------------------------------------------
__global__ void __launch_bounds__(256)
clahe_apply_kernel(const float* __restrict__ x,
                   const unsigned char* __restrict__ lut,
                   float* __restrict__ out, int n4)
{
    int idx = blockIdx.x * blockDim.x + threadIdx.x;
    if (idx >= n4) return;

    const int p4  = idx * 4;
    const int bc  = p4 / IMG_HW;
    const int rem = p4 - bc * IMG_HW;
    const int y   = rem >> 9;
    const int x0  = rem & 511;

    // y interpolation coords (reference: f = y/64 - 0.5, floor before clamp)
    float fy = (float)y * (1.f / 64.f) - 0.5f;
    int   iy = (int)floorf(fy);
    float wy = fy - (float)iy;
    int ty2 = iy + 1; if (ty2 > 7) ty2 = 7;
    int ty1 = iy;     if (ty1 < 0) ty1 = 0;
    const float wy1 = 1.f - wy, wy2 = wy;

    const unsigned char* lbase = lut + bc * (64 * BINS);
    const unsigned char* lr1 = lbase + (ty1 * 8) * BINS;
    const unsigned char* lr2 = lbase + (ty2 * 8) * BINS;

    const float4 xv = *(const float4*)(x + p4);
    float4 ov;
    const float* xi = (const float*)&xv;
    float* oi = (float*)&ov;

    #pragma unroll
    for (int j = 0; j < 4; ++j) {
        int xx = x0 + j;
        float fx = (float)xx * (1.f / 64.f) - 0.5f;
        int   ix = (int)floorf(fx);
        float wx = fx - (float)ix;
        int tx2 = ix + 1; if (tx2 > 7) tx2 = 7;
        int tx1 = ix;     if (tx1 < 0) tx1 = 0;
        float wx1 = 1.f - wx, wx2 = wx;

        int v = (int)rintf(fminf(fmaxf(xi[j], 0.f), 1.f) * 255.f);

        float l11 = (float)lr1[tx1 * BINS + v];
        float l12 = (float)lr1[tx2 * BINS + v];
        float l21 = (float)lr2[tx1 * BINS + v];
        float l22 = (float)lr2[tx2 * BINS + v];

        float r = l11 * (wy1 * wx1) + l12 * (wy1 * wx2)
                + l21 * (wy2 * wx1) + l22 * (wy2 * wx2);
        r = fminf(fmaxf(rintf(r), 0.f), 255.f);
        oi[j] = r / 255.0f;
    }

    *(float4*)(out + p4) = ov;
}

// ---------------------------------------------------------------------------
extern "C" void kernel_launch(void* const* d_in, const int* in_sizes, int n_in,
                              void* d_out, int out_size, void* d_ws, size_t ws_size,
                              hipStream_t stream)
{
    (void)in_sizes; (void)n_in; (void)out_size; (void)ws_size;
    const float* x = (const float*)d_in[0];
    float* out = (float*)d_out;
    unsigned char* lut = (unsigned char*)d_ws;   // 6144*256 = 1.5 MB

    const int n_tiles = 32 * 3 * 8 * 8;          // 6144
    clahe_lut_kernel<<<n_tiles, 256, 0, stream>>>(x, lut);

    const int n4 = 32 * 3 * IMG_HW / 4;          // 6,291,456
    clahe_apply_kernel<<<(n4 + 255) / 256, 256, 0, stream>>>(x, lut, out, n4);
}
